// DeepGCN_60902636257282
// MI455X (gfx1250) — compile-verified, matrix operand order checked by reading, not run on hardware
//
#include <hip/hip_runtime.h>
#include <hip/hip_bf16.h>

// ---------------------------------------------------------------------------
// DeepGCN forward on gfx1250 (MI455X).
//  - GEMMs: V_WMMA_F32_16X16X32_BF16 (bf16 in, fp32 accumulate), weights
//    staged per-block into LDS by the Tensor Data Mover (tensor_load_to_lds,
//    TENSORcnt-tracked), B fragments read via ds_load_b128.
//  - SpMM: global_atomic_add_f32 into an L2-resident 25.6MB output.
//  - BatchNorm/PairNorm: one-pass column sum/sumsq reductions.
//  - gc_b bias skipped: PairNorm column-centering cancels it exactly.
// ---------------------------------------------------------------------------

typedef __bf16 bf16_t;
typedef __attribute__((ext_vector_type(8)))  __bf16 v8bf;
typedef __attribute__((ext_vector_type(16))) __bf16 v16bf;
typedef __attribute__((ext_vector_type(8)))  float  v8f;
typedef unsigned int u32x4 __attribute__((ext_vector_type(4)));
typedef int          i32x4 __attribute__((ext_vector_type(4)));
typedef int          i32x8 __attribute__((ext_vector_type(8)));

#define GCN_N 50000
#define GCN_E 800000
#define GCN_D 128
#define GCN_C 40
#define GCN_L 4

// ---------------------------------------------------------------------------
// Pack a [128 x srcN] row-major fp32 weight matrix into WMMA B-fragment order:
// Wfrag[kt][nt][lane][e] (bf16). B layout (16-bit, 32x16): lane<16 -> N=lane,
// elems 0..7 => K=kt*32+0..7, elems 8..15 => K=kt*32+16..23; lane>=16 ->
// N=lane-16, K offsets +8. Columns >= srcN zero-padded.
// ---------------------------------------------------------------------------
__global__ void pack_wfrag_kernel(const float* __restrict__ W,
                                  bf16_t* __restrict__ Wfrag,
                                  int srcN, int NT) {
    int idx = blockIdx.x * blockDim.x + threadIdx.x;
    int total = 4 * NT * 32 * 16;
    if (idx >= total) return;
    int e    = idx & 15;
    int lane = (idx >> 4) & 31;
    int rest = idx >> 9;           // kt*NT + nt
    int nt   = rest % NT;
    int kt   = rest / NT;
    int hf   = lane >> 4;
    int n    = nt * 16 + (lane & 15);
    int j    = e & 7;
    int khi  = e >> 3;
    int k    = kt * 32 + hf * 8 + khi * 16 + j;
    float v  = (n < srcN) ? W[k * srcN + n] : 0.0f;
    Wfrag[idx] = (bf16_t)v;
}

// ---------------------------------------------------------------------------
// Column sum / sum-of-squares over [N x 128]; stats pre-zeroed.
// ---------------------------------------------------------------------------
__global__ void col_stats_kernel(const float* __restrict__ X,
                                 float* __restrict__ stats, int N) {
    int col        = threadIdx.x & 127;
    int rowsPerBlk = blockDim.x >> 7;
    int r          = blockIdx.x * rowsPerBlk + (threadIdx.x >> 7);
    int rstride    = gridDim.x * rowsPerBlk;
    float s = 0.0f, q = 0.0f;
    for (; r < N; r += rstride) {
        float v = X[r * GCN_D + col];
        s += v;
        q += v * v;
    }
    atomicAdd(&stats[col], s);
    atomicAdd(&stats[128 + col], q);
}

// ---------------------------------------------------------------------------
// BatchNorm1d (batch stats) + affine, emit bf16 activations.
// ---------------------------------------------------------------------------
__global__ void bn_apply_kernel(const float* __restrict__ x,
                                const float* __restrict__ stats,
                                const float* __restrict__ gamma,
                                const float* __restrict__ beta,
                                bf16_t* __restrict__ Xb, int N) {
    int i = blockIdx.x * blockDim.x + threadIdx.x;
    if (i >= N * GCN_D) return;
    int d = i & 127;
    float invN = 1.0f / (float)N;
    float mu   = stats[d] * invN;
    float var  = stats[128 + d] * invN - mu * mu;
    float rstd = rsqrtf(var + 1e-5f);
    Xb[i] = (bf16_t)((x[i] - mu) * rstd * gamma[d] + beta[d]);
}

// ---------------------------------------------------------------------------
// Stage NT*4KB of packed weights into LDS.
// Preferred path: one TDM tensor_load_to_lds issued by wave 0 (EXEC-ignored,
// TENSORcnt-tracked; D# per CDNA5 ISA, 1-D tile).
// This toolchain exposes the 6-arg builtin:
//   (u32x4 g0, i32x8 g1, i32x4 g2, i32x4 g3, i32x8 extra, i32 cpol)
// ---------------------------------------------------------------------------
template <int NT>
__device__ inline void stage_weights_lds(bf16_t* lwf, const bf16_t* Wfrag) {
    constexpr unsigned NELEM = 4u * NT * 512u;         // bf16 elements
#if __has_builtin(__builtin_amdgcn_tensor_load_to_lds)
    if ((threadIdx.x >> 5) == 0) {
        unsigned long long ga = (unsigned long long)Wfrag;
        unsigned lds = (unsigned)(unsigned long long)(uintptr_t)lwf;
        u32x4 g0;
        g0[0] = 1u;                                    // count=1, normal mode
        g0[1] = lds;                                   // lds_addr
        g0[2] = (unsigned)ga;                          // global_addr[31:0]
        g0[3] = (unsigned)((ga >> 32) & 0x1FFFFFFu)    // global_addr[56:32]
                | 0x80000000u;                         // type=2 ("image")
        i32x8 g1;
        g1[0] = (int)(1u << 16);                       // data_size=1 (2B), no mcast
        g1[1] = (int)((NELEM & 0xFFFFu) << 16);        // tensor_dim0[15:0]
        g1[2] = (int)((NELEM >> 16) | (1u << 16));     // tensor_dim0[31:16], dim1=1
        g1[3] = (int)((NELEM & 0xFFFFu) << 16);        // tile_dim0 = NELEM (1-D)
        g1[4] = 0;                                     // tile_dim1=0 (unused), tile_dim2=0
        g1[5] = (int)NELEM;                            // tensor_dim0_stride[31:0]
        g1[6] = 0;
        g1[7] = 0;
        i32x4 z4 = {0, 0, 0, 0};
        i32x8 z8 = {0, 0, 0, 0, 0, 0, 0, 0};
        __builtin_amdgcn_tensor_load_to_lds(g0, g1, z4, z4, z8, 0);
        __builtin_amdgcn_s_wait_tensorcnt(0);
    }
#else
    // Cooperative fallback: 16B per thread per step.
    const v8bf* src = (const v8bf*)Wfrag;
    v8bf* dst = (v8bf*)lwf;
    for (unsigned i = threadIdx.x; i < NELEM / 8; i += blockDim.x)
        dst[i] = src[i];
#endif
    __syncthreads();
}

// ---------------------------------------------------------------------------
// WMMA GEMM: out[M x Ncols] = A[M x 128](bf16) * W(frag in LDS) (+ bias).
// One wave per 16-row M tile; K = 128 = 4 chunks of 32; NT col-tiles of 16.
// A layout (16-bit, 16x32): lane<16 -> M=lane, elems 0..7 => K=koff+0..7,
// elems 8..15 => K=koff+16..23; lane>=16 -> K offsets +8.
// C/D layout: lane half 0 -> rows mBase+0..7 in c[0..7], half 1 -> +8.
// ---------------------------------------------------------------------------
template <int NT, bool STOREBF, bool HASBIAS, bool GUARD>
__global__ void wmma_gemm_kernel(const bf16_t* __restrict__ A,
                                 const bf16_t* __restrict__ Wfrag,
                                 float* __restrict__ outF,
                                 bf16_t* __restrict__ outB,
                                 const float* __restrict__ bias,
                                 int M, int Ncols, int ldo) {
    __shared__ bf16_t lwf[4 * NT * 512];
    stage_weights_lds<NT>(lwf, Wfrag);      // barrier inside (before any exit)

    int wave  = (int)((blockIdx.x * blockDim.x + threadIdx.x) >> 5);
    int lane  = threadIdx.x & 31;
    int mBase = wave * 16;
    if (mBase >= M) return;                 // wave-uniform: EXEC stays all-ones
    int hf   = lane >> 4;
    int l15  = lane & 15;
    int rowA = mBase + l15;

    v16bf a[4];
#pragma unroll
    for (int kt = 0; kt < 4; ++kt) {
        int koff = kt * 32 + hf * 8;
        v8bf lo = *(const v8bf*)(A + rowA * GCN_D + koff);
        v8bf hi = *(const v8bf*)(A + rowA * GCN_D + koff + 16);
        a[kt] = __builtin_shufflevector(lo, hi, 0, 1, 2, 3, 4, 5, 6, 7,
                                                8, 9, 10, 11, 12, 13, 14, 15);
    }

#pragma unroll
    for (int nt = 0; nt < NT; ++nt) {
        v8f c = {};
#pragma unroll
        for (int kt = 0; kt < 4; ++kt) {
            const bf16_t* bp = lwf + (((kt * NT) + nt) * 32 + lane) * 16;
            v8bf blo = *(const v8bf*)(bp);
            v8bf bhi = *(const v8bf*)(bp + 8);
            v16bf b = __builtin_shufflevector(blo, bhi, 0, 1, 2, 3, 4, 5, 6, 7,
                                              8, 9, 10, 11, 12, 13, 14, 15);
            c = __builtin_amdgcn_wmma_f32_16x16x32_bf16(
                    false, a[kt], false, b, (short)0, c, false, false);
        }
        int col = nt * 16 + l15;
        float bv = 0.0f;
        if (HASBIAS) bv = (GUARD && col >= Ncols) ? 0.0f : bias[col];
        if (!GUARD || col < Ncols) {
#pragma unroll
            for (int r = 0; r < 8; ++r) {
                int row = mBase + hf * 8 + r;
                float v = c[r] + bv;
                if (STOREBF) outB[row * ldo + col] = (bf16_t)v;
                else         outF[row * ldo + col] = v;
            }
        }
    }
}

// ---------------------------------------------------------------------------
// SpMM: S[row[e]] += val[e] * H[col[e]]   (one wave per edge, 4 floats/lane).
// ---------------------------------------------------------------------------
__global__ void spmm_kernel(const int* __restrict__ erow,
                            const int* __restrict__ ecol,
                            const float* __restrict__ eval,
                            const float* __restrict__ H,
                            float* __restrict__ S, int E) {
    long long gid = (long long)blockIdx.x * blockDim.x + threadIdx.x;
    int e    = (int)(gid >> 5);
    int lane = (int)(gid & 31);
    if (e >= E) return;
    int r = erow[e];
    int c = ecol[e];
    float v = eval[e];
    const float4 h4 = *((const float4*)(H + c * GCN_D) + lane);
    float* o = S + r * GCN_D + lane * 4;
    atomicAdd(o + 0, v * h4.x);
    atomicAdd(o + 1, v * h4.y);
    atomicAdd(o + 2, v * h4.z);
    atomicAdd(o + 3, v * h4.w);
}

// ---------------------------------------------------------------------------
// PairNorm finalize: stats[d] <- column mean; stats[128] <- rsqrt(eps + sum var).
// ---------------------------------------------------------------------------
__global__ void pn_finalize_kernel(float* __restrict__ stats, int N) {
    __shared__ float red[128];
    int d = threadIdx.x;
    float invN = 1.0f / (float)N;
    float mu  = stats[d] * invN;
    float var = stats[128 + d] * invN - mu * mu;
    stats[d] = mu;
    red[d] = var;
    __syncthreads();
    for (int o = 64; o > 0; o >>= 1) {
        if (d < o) red[d] += red[d + o];
        __syncthreads();
    }
    if (d == 0) stats[128] = rsqrtf(1e-6f + red[0]);   // PN_SCALE == 1.0
}

// ---------------------------------------------------------------------------
// PairNorm apply + ReLU + residual; emits fp32 state and bf16 for next GEMM.
// ---------------------------------------------------------------------------
__global__ void pn_apply_kernel(const float* __restrict__ S,
                                const float* __restrict__ stats,
                                float* __restrict__ CUR,
                                bf16_t* __restrict__ Xb,
                                int N, int addPrev) {
    int i = blockIdx.x * blockDim.x + threadIdx.x;
    if (i >= N * GCN_D) return;
    int d = i & 127;
    float invr = stats[128];
    float v = (S[i] - stats[d]) * invr;
    v = fmaxf(v, 0.0f);
    if (addPrev) v += CUR[i];
    CUR[i] = v;
    Xb[i] = (bf16_t)v;
}

// ---------------------------------------------------------------------------
// Host-side launch. Inputs (setup_inputs order):
// 0 x, 1 edge_row, 2 edge_col, 3 edge_val, 4 bn_gamma, 5 bn_beta,
// 6 fc_in_w, 7 fc_in_b, 8 gc_w, 9 gc_b (cancelled by PairNorm),
// 10 fc_out_w, 11 fc_out_b.  d_out: [N*C] f32.
// ---------------------------------------------------------------------------
extern "C" void kernel_launch(void* const* d_in, const int* in_sizes, int n_in,
                              void* d_out, int out_size, void* d_ws, size_t ws_size,
                              hipStream_t stream) {
    const float* x        = (const float*)d_in[0];
    const int*   erow     = (const int*)d_in[1];
    const int*   ecol     = (const int*)d_in[2];
    const float* eval_    = (const float*)d_in[3];
    const float* bn_gamma = (const float*)d_in[4];
    const float* bn_beta  = (const float*)d_in[5];
    const float* fc_in_w  = (const float*)d_in[6];
    const float* fc_in_b  = (const float*)d_in[7];
    const float* gc_w     = (const float*)d_in[8];
    const float* fc_out_w = (const float*)d_in[10];
    const float* fc_out_b = (const float*)d_in[11];
    float* out = (float*)d_out;

    // -------- workspace layout --------
    char* ws = (char*)d_ws;
    size_t off = 0;
    float*  stats  = (float*)(ws + off);  off += 1024;                        // 256 f32
    bf16_t* wf_in  = (bf16_t*)(ws + off); off += 4 * 8 * 512 * 2;             // 32KB
    bf16_t* wf_gc  = (bf16_t*)(ws + off); off += (size_t)4 * 4 * 8 * 512 * 2; // 128KB
    bf16_t* wf_out = (bf16_t*)(ws + off); off += 4 * 3 * 512 * 2;             // 12KB
    off = (off + 255) & ~(size_t)255;
    bf16_t* XB2 = (bf16_t*)(ws + off); off += (size_t)GCN_N * GCN_D * 2;
    bf16_t* XB  = (bf16_t*)(ws + off); off += (size_t)GCN_N * GCN_D * 2;
    float*  H   = (float*)(ws + off);  off += (size_t)GCN_N * GCN_D * 4;
    float*  S   = (float*)(ws + off);  off += (size_t)GCN_N * GCN_D * 4;
    float*  CUR = (float*)(ws + off);  off += (size_t)GCN_N * GCN_D * 4;

    const int ND        = GCN_N * GCN_D;
    const int ELEM_BLK  = 256;
    const int elemGrid  = (ND + ELEM_BLK - 1) / ELEM_BLK;
    const int gemmWaves = GCN_N / 16;                        // 3125
    const int gemmGrid  = (gemmWaves + 7) / 8;               // 8 waves/block
    const int spmmGrid  = (int)(((long long)GCN_E * 32 + 255) / 256);

    // -------- weight packing (tiny) --------
    pack_wfrag_kernel<<<(4 * 8 * 512 + 255) / 256, 256, 0, stream>>>(fc_in_w, wf_in, GCN_D, 8);
    for (int l = 0; l < GCN_L; ++l)
        pack_wfrag_kernel<<<(4 * 8 * 512 + 255) / 256, 256, 0, stream>>>(
            gc_w + (size_t)l * GCN_D * GCN_D, wf_gc + (size_t)l * 4 * 8 * 512, GCN_D, 8);
    pack_wfrag_kernel<<<(4 * 3 * 512 + 255) / 256, 256, 0, stream>>>(fc_out_w, wf_out, GCN_C, 3);

    // -------- BatchNorm --------
    (void)hipMemsetAsync(stats, 0, 1024, stream);
    col_stats_kernel<<<512, 256, 0, stream>>>(x, stats, GCN_N);
    bn_apply_kernel<<<elemGrid, ELEM_BLK, 0, stream>>>(x, stats, bn_gamma, bn_beta, XB2, GCN_N);

    // -------- fc_in: XB = bf16(BN(x) @ Win + bin) --------
    wmma_gemm_kernel<8, true, true, false><<<gemmGrid, 256, 0, stream>>>(
        XB2, wf_in, nullptr, XB, fc_in_b, GCN_N, GCN_D, GCN_D);

    // -------- GCN layers --------
    for (int l = 0; l < GCN_L; ++l) {
        wmma_gemm_kernel<8, false, false, false><<<gemmGrid, 256, 0, stream>>>(
            XB, wf_gc + (size_t)l * 4 * 8 * 512, H, nullptr, nullptr, GCN_N, GCN_D, GCN_D);
        (void)hipMemsetAsync(S, 0, (size_t)ND * 4, stream);
        spmm_kernel<<<spmmGrid, 256, 0, stream>>>(erow, ecol, eval_, H, S, GCN_E);
        (void)hipMemsetAsync(stats, 0, 1024, stream);
        col_stats_kernel<<<512, 256, 0, stream>>>(S, stats, GCN_N);
        pn_finalize_kernel<<<1, 128, 0, stream>>>(stats, GCN_N);
        pn_apply_kernel<<<elemGrid, ELEM_BLK, 0, stream>>>(S, stats, CUR, XB, GCN_N,
                                                           (l > 0) ? 1 : 0);
    }

    // -------- fc_out: out = x @ Wout + bout --------
    wmma_gemm_kernel<3, false, true, true><<<gemmGrid, 256, 0, stream>>>(
        XB, wf_out, out, nullptr, fc_out_b, GCN_N, GCN_C, GCN_C);
}
